// SwinBlock_15934328669044
// MI455X (gfx1250) — matrix operand order repaired and checked by reading, not compiled, not run on hardware
//
#include <hip/hip_runtime.h>
#include <hip/hip_bf16.h>
#include <math.h>

// Fused Swin block for MI455X (gfx1250, wave32, WMMA).
// One workgroup (256 threads = 8 waves) processes one 7x7 window entirely in LDS.
// bf16 WMMA matmuls with f32 accumulation; LN / softmax / GELU / residual in f32.
// Stage-0 input staging uses GLOBAL_LOAD_ASYNC_TO_LDS_B128 (ASYNCcnt path).
// Weights converted once per launch to bf16, transposed to [out][in] in d_ws
// (~3.4 MB, stays L2-resident for all 1024 blocks).

typedef __attribute__((ext_vector_type(16))) __bf16        v16bf;
typedef __attribute__((ext_vector_type(8)))  float         v8f;
typedef __attribute__((ext_vector_type(4)))  unsigned int  u32x4;

#define WSZ   7
#define NTOK  49
#define MPAD  64
#define CDIM  384
#define C3    1152
#define CHID  1536
#define HSTR  400   // bf16 halves per row (384 + 16 pad), 800B = 16B-aligned
#define QSTR  400
#define VSTR  72    // V^T: [384][72] tokens padded
#define SSTR  68    // f32 scores stride
#define ATSTR 72    // bf16 attn stride
#define USTR  136   // MLP chunk: [64][136] (128 + 8 pad)

// LDS layout (bytes). Total 307,200 <= 320KB per WGP workgroup.
#define OFF_X   0                       // f32  [64][384]  = 98304
#define OFF_H   98304                   // bf16 [64][400]  = 51200 (LN1 out, later LN2 out)
#define OFF_Q   149504                  // bf16 [64][400]  = 51200 (Q, later attn-out O)
#define OFF_K   200704                  // bf16 [64][400]  = 51200 (K, later MLP chunk U)
#define OFF_V   251904                  // bf16 [384][72]  = 55296 (V transposed)
#define OFF_S   OFF_H                   // f32  [64][68]   = 17408 (aliases dead H)
#define OFF_ATT (OFF_H + 17408)         // bf16 [64][72]   = 9216
#define OFF_U   OFF_K                   // bf16 [64][136]  = 17408 (aliases dead K)
#define LDS_BYTES 307200

// ---- WMMA fragment loaders (contiguous k = 16*hi + s in A and B) ------------------------------
// A fragment, source row-major [M][K]: lane row = lane%16, 32 contiguous
// bytes of K starting at hi*16 (hi = lane>>4).
__device__ inline v16bf ld_a(const __bf16* m, int stride, int row, int kbase, int hi) {
    const __bf16* r = m + (size_t)row * stride + kbase + hi * 16;
    union { u32x4 q[2]; v16bf v; } u;
    u.q[0] = *(const u32x4*)(r);
    u.q[1] = *(const u32x4*)(r + 8);
    return u.v;
}
// B fragment, source [N][K] ("transposed weight"): lane row = N = lane%16,
// 32 contiguous bytes of K starting at hi*16.
__device__ inline v16bf ld_b(const __bf16* m, int stride, int row, int kbase, int hi) {
    const __bf16* r = m + (size_t)row * stride + kbase + hi * 16;
    union { u32x4 q[2]; v16bf v; } u;
    u.q[0] = *(const u32x4*)(r);
    u.q[1] = *(const u32x4*)(r + 8);
    return u.v;
}
__device__ inline v8f wmma_bf16(v16bf a, v16bf b, v8f c) {
    return __builtin_amdgcn_wmma_f32_16x16x32_bf16(false, a, false, b, (short)0, c,
                                                   false, false);
}

__device__ inline float wave_sum(float s) {
#pragma unroll
    for (int o = 16; o > 0; o >>= 1) s += __shfl_xor(s, o, 32);
    return s;
}
__device__ inline float wave_max(float s) {
#pragma unroll
    for (int o = 16; o > 0; o >>= 1) s = fmaxf(s, __shfl_xor(s, o, 32));
    return s;
}

// Async global->LDS 16B copy (GLOBAL_LOAD_ASYNC_TO_LDS_B128, ASYNCcnt-tracked).
// lds_off: workgroup-relative LDS byte offset (low 32 bits of generic LDS ptr).
__device__ inline void async_g2l_b128(unsigned lds_off, const void* gsrc) {
    unsigned long long ga = (unsigned long long)(size_t)gsrc;
    asm volatile("global_load_async_to_lds_b128 %0, %1, off"
                 :: "v"(lds_off), "v"(ga)
                 : "memory");
}
__device__ inline void wait_async0() {
    asm volatile("s_wait_asynccnt 0x0" ::: "memory");
}

// ---- weight convert + transpose: f32 [in][out] -> bf16 [out][in] -------------
__global__ void conv_transpose_bf16(const float* __restrict__ src,
                                    __bf16* __restrict__ dst,
                                    int in_dim, int out_dim) {
    int idx = blockIdx.x * blockDim.x + threadIdx.x;
    if (idx >= in_dim * out_dim) return;
    int o = idx / in_dim;
    int i = idx - o * in_dim;
    dst[idx] = (__bf16)src[(size_t)i * out_dim + o];
}

// ---- fused Swin block --------------------------------------------------------
__global__ void __launch_bounds__(256)
swin_block_kernel(const float* __restrict__ x,
                  const float* __restrict__ g1, const float* __restrict__ b1,
                  const float* __restrict__ bqkv, const float* __restrict__ bp,
                  const float* __restrict__ g2,  const float* __restrict__ b2,
                  const float* __restrict__ bm1, const float* __restrict__ bm2,
                  const __bf16* __restrict__ WqkvT, const __bf16* __restrict__ WpT,
                  const __bf16* __restrict__ Wm1T,  const __bf16* __restrict__ Wm2T,
                  float* __restrict__ out) {
    extern __shared__ char smem[];
    float*  sX = (float*)(smem + OFF_X);
    __bf16* sH = (__bf16*)(smem + OFF_H);
    __bf16* sQ = (__bf16*)(smem + OFF_Q);
    __bf16* sK = (__bf16*)(smem + OFF_K);
    __bf16* sV = (__bf16*)(smem + OFF_V);
    float*  sS = (float*)(smem + OFF_S);
    __bf16* sA = (__bf16*)(smem + OFF_ATT);
    __bf16* sU = (__bf16*)(smem + OFF_U);

    const int tid  = threadIdx.x;
    const int wv   = tid >> 5;
    const int lane = tid & 31;
    const int l15  = lane & 15;
    const int hi   = lane >> 4;

    const int widx = blockIdx.x;           // 1024 windows
    const int b  = widx >> 6;
    const int wi = widx & 63;
    const int wh = wi >> 3;
    const int ww = wi & 7;

    if (tid == 0) {                        // warm L2 path for the shared weights
        __builtin_prefetch(WqkvT, 0, 0);
        __builtin_prefetch(Wm1T, 0, 0);
        __builtin_prefetch(Wm2T, 0, 0);
    }

    // ---- Stage 0a: async-stage the 49 window rows of x straight into LDS ----
    {
        const int chunks_per_row = CDIM / 4;               // 96 x 16B per row
        const int total = NTOK * chunks_per_row;           // 4704 b128 copies
        for (int i = tid; i < total; i += 256) {
            int row = i / chunks_per_row;
            int ch4 = i - row * chunks_per_row;
            int tr = row / WSZ, tc = row % WSZ;
            int n  = (wh * WSZ + tr) * 56 + (ww * WSZ + tc);
            const float* gsrc = x + ((size_t)(b * 3136 + n)) * CDIM + ch4 * 4;
            unsigned lds_off = (unsigned)(size_t)(sX + (size_t)row * CDIM + ch4 * 4);
            async_g2l_b128(lds_off, gsrc);
        }
        // zero the 15 pad rows with plain LDS stores (overlaps the async copies)
        for (int i = tid; i < (MPAD - NTOK) * CDIM; i += 256) {
            int row = NTOK + i / CDIM, c = i - (i / CDIM) * CDIM;
            sX[(size_t)row * CDIM + c] = 0.f;
        }
        wait_async0();
    }
    __syncthreads();

    // ---- Stage 0b: LayerNorm1 from LDS (one wave per row) ----
    for (int rr = 0; rr < 8; ++rr) {
        int row = wv * 8 + rr;
        float xv[12];
#pragma unroll
        for (int j = 0; j < 12; ++j) xv[j] = sX[(size_t)row * CDIM + lane + j * 32];
        float s = 0.f;
#pragma unroll
        for (int j = 0; j < 12; ++j) s += xv[j];
        float mu = wave_sum(s) * (1.f / CDIM);
        float vr = 0.f;
#pragma unroll
        for (int j = 0; j < 12; ++j) { float d = xv[j] - mu; vr += d * d; }
        float rs = rsqrtf(wave_sum(vr) * (1.f / CDIM) + 1e-5f);
#pragma unroll
        for (int j = 0; j < 12; ++j) {
            int c = lane + j * 32;
            float h = (row < NTOK) ? (xv[j] - mu) * rs * g1[c] + b1[c] : 0.f;
            sH[(size_t)row * HSTR + c] = (__bf16)h;
        }
    }
    __syncthreads();

    // ---- Stage 1: QKV = h @ Wqkv + bqkv   (64x1152x384, 288 tiles / 8 waves) ----
    for (int t = 0; t < 36; ++t) {
        int g  = wv * 36 + t;
        int mt = g / 72, nt = g % 72;
        int arow = mt * 16 + l15;
        int brow = nt * 16 + l15;
        v8f d = {};
#pragma unroll
        for (int ks = 0; ks < 12; ++ks) {
            v16bf a  = ld_a(sH, HSTR, arow, ks * 32, hi);
            v16bf bb = ld_b(WqkvT, CDIM, brow, ks * 32, hi);
            d = wmma_bf16(a, bb, d);
        }
        float bias = bqkv[nt * 16 + l15];
        int mbase = mt * 16 + hi * 8;
        int col   = nt * 16 + l15;
#pragma unroll
        for (int r = 0; r < 8; ++r) {
            float val = d[r] + bias;
            if (nt < 24)
                sQ[(size_t)(mbase + r) * QSTR + col] = (__bf16)val;
            else if (nt < 48)
                sK[(size_t)(mbase + r) * QSTR + (col - CDIM)] = (__bf16)val;
            else  // V stored transposed: [channel][token]
                sV[(size_t)(col - 2 * CDIM) * VSTR + (mbase + r)] = (__bf16)val;
        }
    }
    __syncthreads();

    // ---- Stage 2: scores = (q @ k^T) * C^-0.5   (64x64x384) ----
    const float scale = 0.05103103630798288f;  // 384^-0.5
    for (int t = 0; t < 2; ++t) {
        int g  = wv * 2 + t;
        int mt = g >> 2, nt = g & 3;
        v8f d = {};
#pragma unroll
        for (int ks = 0; ks < 12; ++ks) {
            v16bf a  = ld_a(sQ, QSTR, mt * 16 + l15, ks * 32, hi);
            v16bf bb = ld_b(sK, QSTR, nt * 16 + l15, ks * 32, hi);
            d = wmma_bf16(a, bb, d);
        }
        int mbase = mt * 16 + hi * 8, col = nt * 16 + l15;
#pragma unroll
        for (int r = 0; r < 8; ++r)
            sS[(size_t)(mbase + r) * SSTR + col] = d[r] * scale;
    }
    __syncthreads();

    // ---- Stage 3: row softmax (mask cols >= 49), attn -> bf16 ----
    for (int rr = 0; rr < 8; ++rr) {
        int row = wv * 8 + rr;
        float v0 = (lane < NTOK)      ? sS[(size_t)row * SSTR + lane]      : -INFINITY;
        float v1 = (lane + 32 < NTOK) ? sS[(size_t)row * SSTR + lane + 32] : -INFINITY;
        float m  = wave_max(fmaxf(v0, v1));
        float e0 = (lane < NTOK)      ? __expf(v0 - m) : 0.f;
        float e1 = (lane + 32 < NTOK) ? __expf(v1 - m) : 0.f;
        float sm = wave_sum(e0 + e1);
        float inv = (row < NTOK) ? 1.f / sm : 0.f;   // pad rows -> zeros
        sA[(size_t)row * ATSTR + lane]      = (__bf16)(e0 * inv);
        sA[(size_t)row * ATSTR + lane + 32] = (__bf16)(e1 * inv);
    }
    __syncthreads();

    // ---- Stage 4: O = attn @ V   (64x384x64), store into sQ (Q is dead) ----
    for (int t = 0; t < 12; ++t) {
        int g  = wv * 12 + t;
        int mt = g / 24, nt = g % 24;
        v8f d = {};
#pragma unroll
        for (int ks = 0; ks < 2; ++ks) {
            v16bf a  = ld_a(sA, ATSTR, mt * 16 + l15, ks * 32, hi);
            v16bf bb = ld_b(sV, VSTR, nt * 16 + l15, ks * 32, hi);
            d = wmma_bf16(a, bb, d);
        }
        int mbase = mt * 16 + hi * 8, col = nt * 16 + l15;
#pragma unroll
        for (int r = 0; r < 8; ++r)
            sQ[(size_t)(mbase + r) * QSTR + col] = (__bf16)d[r];
    }
    __syncthreads();

    // ---- Stage 5: y = x + O @ Wp + bp  (64x384x384), in-place into sX ----
    for (int t = 0; t < 12; ++t) {
        int g  = wv * 12 + t;
        int mt = g / 24, nt = g % 24;
        v8f d = {};
#pragma unroll
        for (int ks = 0; ks < 12; ++ks) {
            v16bf a  = ld_a(sQ, QSTR, mt * 16 + l15, ks * 32, hi);
            v16bf bb = ld_b(WpT, CDIM, nt * 16 + l15, ks * 32, hi);
            d = wmma_bf16(a, bb, d);
        }
        int mbase = mt * 16 + hi * 8, col = nt * 16 + l15;
        float bias = bp[col];
#pragma unroll
        for (int r = 0; r < 8; ++r) {
            size_t off = (size_t)(mbase + r) * CDIM + col;
            sX[off] = sX[off] + d[r] + bias;
        }
    }
    __syncthreads();

    // ---- Stage 6: LayerNorm2(y) -> sH (overwrites dead scores/attn region) ----
    for (int rr = 0; rr < 8; ++rr) {
        int row = wv * 8 + rr;
        float xv[12];
#pragma unroll
        for (int j = 0; j < 12; ++j) xv[j] = sX[(size_t)row * CDIM + lane + j * 32];
        float s = 0.f;
#pragma unroll
        for (int j = 0; j < 12; ++j) s += xv[j];
        float mu = wave_sum(s) * (1.f / CDIM);
        float vr = 0.f;
#pragma unroll
        for (int j = 0; j < 12; ++j) { float d = xv[j] - mu; vr += d * d; }
        float rs = rsqrtf(wave_sum(vr) * (1.f / CDIM) + 1e-5f);
#pragma unroll
        for (int j = 0; j < 12; ++j) {
            int c = lane + j * 32;
            float h = (row < NTOK) ? (xv[j] - mu) * rs * g2[c] + b2[c] : 0.f;
            sH[(size_t)row * HSTR + c] = (__bf16)h;
        }
    }
    __syncthreads();

    // ---- Stage 7: MLP, chunked over 1536 hidden in 12 x 128 columns ----
    // Each wave owns 12 persistent output tiles of the 64x384 result (96 VGPRs).
    const v8f vzero = {};
    v8f acc[12];
#pragma unroll
    for (int i = 0; i < 12; ++i) acc[i] = vzero;

    for (int cc = 0; cc < 12; ++cc) {
        // 7a: U = gelu(h2 @ Wm1[:,chunk] + bm1[chunk]) -> sU (bf16)
        for (int t = 0; t < 4; ++t) {
            int g  = wv * 4 + t;
            int mt = g >> 3, nt = g & 7;
            v8f d = {};
#pragma unroll
            for (int ks = 0; ks < 12; ++ks) {
                v16bf a  = ld_a(sH, HSTR, mt * 16 + l15, ks * 32, hi);
                v16bf bb = ld_b(Wm1T, CDIM, cc * 128 + nt * 16 + l15, ks * 32, hi);
                d = wmma_bf16(a, bb, d);
            }
            int mbase = mt * 16 + hi * 8, col = nt * 16 + l15;
            float bias = bm1[cc * 128 + col];
#pragma unroll
            for (int r = 0; r < 8; ++r) {
                float u = d[r] + bias;
                u = 0.5f * u * (1.f + erff(u * 0.70710678118654752f));  // exact gelu
                sU[(size_t)(mbase + r) * USTR + col] = (__bf16)u;
            }
        }
        __syncthreads();

        // 7b: acc += U @ Wm2[chunk,:]  (K = 128 per chunk)
#pragma unroll
        for (int i = 0; i < 12; ++i) {
            int g  = wv * 12 + i;
            int mt = g / 24, nt = g % 24;
#pragma unroll
            for (int ks = 0; ks < 4; ++ks) {
                v16bf a  = ld_a(sU, USTR, mt * 16 + l15, ks * 32, hi);
                v16bf bb = ld_b(Wm2T, CHID, nt * 16 + l15, cc * 128 + ks * 32, hi);
                acc[i] = wmma_bf16(a, bb, acc[i]);
            }
        }
        __syncthreads();  // before next chunk overwrites sU
    }

    // ---- Stage 8: out = y + acc + bm2, scattered back to (B, N, C) layout ----
#pragma unroll
    for (int i = 0; i < 12; ++i) {
        int g  = wv * 12 + i;
        int mt = g / 24, nt = g % 24;
        int mbase = mt * 16 + hi * 8, col = nt * 16 + l15;
        float bias = bm2[col];
#pragma unroll
        for (int r = 0; r < 8; ++r) {
            int row = mbase + r;
            if (row < NTOK) {
                int tr = row / WSZ, tc = row % WSZ;
                int n  = (wh * WSZ + tr) * 56 + (ww * WSZ + tc);
                out[((size_t)(b * 3136 + n)) * CDIM + col] =
                    sX[(size_t)row * CDIM + col] + acc[i][r] + bias;
            }
        }
    }
}

extern "C" void kernel_launch(void* const* d_in, const int* in_sizes, int n_in,
                              void* d_out, int out_size, void* d_ws, size_t ws_size,
                              hipStream_t stream) {
    const float* x    = (const float*)d_in[0];
    const float* g1   = (const float*)d_in[1];
    const float* b1   = (const float*)d_in[2];
    const float* Wqkv = (const float*)d_in[3];
    const float* bqkv = (const float*)d_in[4];
    const float* Wp   = (const float*)d_in[5];
    const float* bp   = (const float*)d_in[6];
    const float* g2   = (const float*)d_in[7];
    const float* b2   = (const float*)d_in[8];
    const float* Wm1  = (const float*)d_in[9];
    const float* bm1  = (const float*)d_in[10];
    const float* Wm2  = (const float*)d_in[11];
    const float* bm2  = (const float*)d_in[12];
    float* out = (float*)d_out;

    // bf16 transposed weights in workspace (3,538,944 bytes total)
    __bf16* WqkvT = (__bf16*)d_ws;
    __bf16* WpT   = WqkvT + (size_t)C3 * CDIM;
    __bf16* Wm1T  = WpT + (size_t)CDIM * CDIM;
    __bf16* Wm2T  = Wm1T + (size_t)CHID * CDIM;

    int nq = CDIM * C3;
    conv_transpose_bf16<<<(nq + 255) / 256, 256, 0, stream>>>(Wqkv, WqkvT, CDIM, C3);
    int np = CDIM * CDIM;
    conv_transpose_bf16<<<(np + 255) / 256, 256, 0, stream>>>(Wp, WpT, CDIM, CDIM);
    int n1 = CDIM * CHID;
    conv_transpose_bf16<<<(n1 + 255) / 256, 256, 0, stream>>>(Wm1, Wm1T, CDIM, CHID);
    conv_transpose_bf16<<<(n1 + 255) / 256, 256, 0, stream>>>(Wm2, Wm2T, CHID, CDIM);

    // 1024 windows, one per workgroup; 307,200 B dynamic LDS (<= 320KB/WGP)
    swin_block_kernel<<<1024, 256, LDS_BYTES, stream>>>(
        x, g1, b1, bqkv, bp, g2, b2, bm1, bm2, WqkvT, WpT, Wm1T, Wm2T, out);
}
